// DecoderLayer_30030411334211
// MI455X (gfx1250) — matrix operand order repaired and checked by reading, not compiled, not run on hardware
//
#include <hip/hip_runtime.h>
#include <hip/hip_bf16.h>

typedef __attribute__((ext_vector_type(16))) _Float16 v16h;
typedef __attribute__((ext_vector_type(8)))  _Float16 v8h;
typedef __attribute__((ext_vector_type(8)))  float    v8f;

#define TB     2
#define TS     2048
#define TDM    512
#define TH     8
#define THID   2048
#define TDEPTH 64

// ---------------------------------------------------------------------------
// Shared WMMA GEMM tile body.
// Block = 128 threads (4 waves). Block tile = 128(M) x 64(N), K stepped by 32.
// Wave w computes rows [32w, 32w+32): 2 A-fragments x 4 B-fragments = 8 WMMA
// per K-step, each B fragment reused twice. All LDS fragment reads are
// contiguous (v8h/v16h typed loads -> ds_load_b128).
// ---------------------------------------------------------------------------
__device__ __forceinline__ void gemm_tile_body(
    const float* __restrict__ A, int lda,
    const float* __restrict__ Bm, int ldb, int transB,
    float* __restrict__ C, int ldc,
    int M, int N, int K,
    const float* __restrict__ bias, int relu,
    float scale, const float* __restrict__ mask, int maskMode,
    int rowBase, int colBase)
{
    // Row stride 48 halfs = 96 B (multiple of 32 B -> aligned vector ops).
    __shared__ __align__(32) _Float16 As[128][48];   // As[m][k]
    __shared__ __align__(32) _Float16 BsT[64][48];   // BsT[n][k]  (transposed!)

    const int tid  = threadIdx.x;     // 0..127
    const int wave = tid >> 5;        // 0..3
    const int lane = tid & 31;
    const int half = lane >> 4;       // lane half selects K sub-range
    const int l15  = lane & 15;

    v8f acc[2][4];
    #pragma unroll
    for (int i = 0; i < 2; ++i)
        #pragma unroll
        for (int j = 0; j < 4; ++j)
            acc[i][j] = (v8f){0.f,0.f,0.f,0.f,0.f,0.f,0.f,0.f};

    for (int kBase = 0; kBase < K; kBase += 32) {
        // ---- A tile 128x32: 256 chunks of 16; each thread fills 2 chunks ----
        #pragma unroll
        for (int cc = 0; cc < 2; ++cc) {
            int li = tid + 128 * cc;
            int r  = li >> 1;
            int c0 = (li & 1) * 16;
            const float* src = A + (size_t)(rowBase + r) * lda + kBase + c0;
            v8h h0, h1;
            #pragma unroll
            for (int j = 0; j < 8; ++j) h0[j] = (_Float16)src[j];
            #pragma unroll
            for (int j = 0; j < 8; ++j) h1[j] = (_Float16)src[8 + j];
            *(v8h*)&As[r][c0]     = h0;
            *(v8h*)&As[r][c0 + 8] = h1;
        }
        // ---- B tile -> BsT[n][k]: 128 chunks of 16; one chunk per thread ----
        {
            int n  = tid >> 1;
            int k0 = (tid & 1) * 16;
            v8h h0, h1;
            if (transB) {
                // math-B[k][n] = Bm[n][k] (row-major): contiguous in k. (QK^T)
                const float* src = Bm + (size_t)(colBase + n) * ldb + kBase + k0;
                #pragma unroll
                for (int j = 0; j < 8; ++j) h0[j] = (_Float16)src[j];
                #pragma unroll
                for (int j = 0; j < 8; ++j) h1[j] = (_Float16)src[8 + j];
            } else {
                // math-B[k][n] row-major: gather down column n (wave-coalesced).
                const float* src = Bm + (size_t)(kBase + k0) * ldb + colBase + n;
                #pragma unroll
                for (int j = 0; j < 8; ++j) h0[j] = (_Float16)src[(size_t)j * ldb];
                #pragma unroll
                for (int j = 0; j < 8; ++j) h1[j] = (_Float16)src[(size_t)(8 + j) * ldb];
            }
            *(v8h*)&BsT[n][k0]     = h0;
            *(v8h*)&BsT[n][k0 + 8] = h1;
        }
        __syncthreads();

        // ---- A fragments (wave32 16x32 f16 layout), contiguous v8h loads ----
        v16h afrag[2];
        #pragma unroll
        for (int ms = 0; ms < 2; ++ms) {
            int r  = wave * 32 + ms * 16 + l15;
            int kb = half * 16;
            v8h lo = *(const v8h*)&As[r][kb];        // K kb..kb+7
            v8h hi = *(const v8h*)&As[r][kb + 8];    // K kb+8..kb+15
            afrag[ms] = __builtin_shufflevector(lo, hi,
                0,1,2,3,4,5,6,7,8,9,10,11,12,13,14,15);
        }
        // ---- 4 B fragments, each one contiguous v16h load; reused twice ----
        #pragma unroll
        for (int ns = 0; ns < 4; ++ns) {
            int c = ns * 16 + l15;                   // lane -> column
            v16h b = *(const v16h*)&BsT[c][half * 16];  // K half*16 .. +15
            acc[0][ns] = __builtin_amdgcn_wmma_f32_16x16x32_f16(
                false, afrag[0], false, b, (short)0, acc[0][ns], false, false);
            acc[1][ns] = __builtin_amdgcn_wmma_f32_16x16x32_f16(
                false, afrag[1], false, b, (short)0, acc[1][ns], false, false);
        }
        __syncthreads();
    }

    // ---- epilogue: C/D layout M = half*8 + vgpr, N = l15 ----
    #pragma unroll
    for (int ms = 0; ms < 2; ++ms) {
        #pragma unroll
        for (int ns = 0; ns < 4; ++ns) {
            int col = colBase + ns * 16 + l15;
            #pragma unroll
            for (int vv = 0; vv < 8; ++vv) {
                int row = rowBase + wave * 32 + ms * 16 + half * 8 + vv;
                if (row < M && col < N) {
                    float val = acc[ms][ns][vv];
                    if (mask) {   // attention-logits epilogue
                        float m = (maskMode == 1) ? mask[(size_t)row * N + col]
                                                  : mask[col];
                        val = val * scale + m * (-1e9f);
                    } else {
                        if (bias) val += bias[col];
                        if (relu) val = val > 0.0f ? val : 0.0f;
                    }
                    C[(size_t)row * ldc + col] = val;
                }
            }
        }
    }
}

// ---------------------------------------------------------------------------
// Wrappers
// ---------------------------------------------------------------------------
__global__ __launch_bounds__(128)
void gemm_bias_kernel(const float* __restrict__ A, int lda,
                      const float* __restrict__ Bm, int ldb,
                      float* __restrict__ C, int ldc,
                      int M, int N, int K,
                      const float* __restrict__ bias, int relu)
{
    gemm_tile_body(A, lda, Bm, ldb, 0, C, ldc, M, N, K, bias, relu,
                   1.0f, nullptr, 0, blockIdx.y * 128, blockIdx.x * 64);
}

__global__ __launch_bounds__(128)
void attn_logits_kernel(const float* __restrict__ Qb,
                        const float* __restrict__ Kb,
                        float* __restrict__ AW,
                        const float* __restrict__ mask, int maskMode)
{
    int z = blockIdx.z;               // b*H + h
    int b = z / TH, h = z % TH;
    const float* Aq = Qb + (size_t)b * TS * TDM + h * TDEPTH;
    const float* Bk = Kb + (size_t)b * TS * TDM + h * TDEPTH;
    float* Cc = AW + (size_t)z * TS * TS;
    gemm_tile_body(Aq, TDM, Bk, TDM, 1, Cc, TS, TS, TS, TDEPTH,
                   nullptr, 0, 0.125f /* 1/sqrt(64) */, mask, maskMode,
                   blockIdx.y * 128, blockIdx.x * 64);
}

__global__ __launch_bounds__(128)
void ctx_gemm_kernel(const float* __restrict__ AW,
                     const float* __restrict__ Vb,
                     float* __restrict__ Ctx)
{
    int z = blockIdx.z;
    int b = z / TH, h = z % TH;
    const float* Aa = AW + (size_t)z * TS * TS;
    const float* Bv = Vb  + (size_t)b * TS * TDM + h * TDEPTH;
    float* Cc       = Ctx + (size_t)b * TS * TDM + h * TDEPTH;
    gemm_tile_body(Aa, TS, Bv, TDM, 0, Cc, TDM, TS, TDEPTH, TS,
                   nullptr, 0, 1.0f, nullptr, 0,
                   blockIdx.y * 128, blockIdx.x * 64);
}

// ---------------------------------------------------------------------------
// Row softmax (in place), one block per row of length TS.
// ---------------------------------------------------------------------------
__global__ __launch_bounds__(256)
void softmax_kernel(float* __restrict__ AW)
{
    __shared__ float red[256];
    float* p = AW + (size_t)blockIdx.x * TS;
    int tid = threadIdx.x;

    float mx = -3.0e38f;
    for (int i = tid; i < TS; i += 256) mx = fmaxf(mx, p[i]);
    red[tid] = mx; __syncthreads();
    for (int s = 128; s > 0; s >>= 1) {
        if (tid < s) red[tid] = fmaxf(red[tid], red[tid + s]);
        __syncthreads();
    }
    mx = red[0]; __syncthreads();

    float sum = 0.0f;
    for (int i = tid; i < TS; i += 256) {
        float e = __expf(p[i] - mx);
        p[i] = e;
        sum += e;
    }
    red[tid] = sum; __syncthreads();
    for (int s = 128; s > 0; s >>= 1) {
        if (tid < s) red[tid] += red[tid + s];
        __syncthreads();
    }
    float inv = 1.0f / red[0];
    for (int i = tid; i < TS; i += 256) p[i] *= inv;
}

// ---------------------------------------------------------------------------
// Residual add + LayerNorm over DM=512, one block per row, 256 threads.
// ---------------------------------------------------------------------------
__global__ __launch_bounds__(256)
void add_ln_kernel(const float* __restrict__ X,
                   const float* __restrict__ Res,
                   const float* __restrict__ g,
                   const float* __restrict__ bta,
                   float* __restrict__ Out)
{
    __shared__ float red[256];
    size_t row = blockIdx.x;
    const float* x = X   + row * TDM;
    const float* r = Res + row * TDM;
    float* o       = Out + row * TDM;
    int tid = threadIdx.x;

    float v0 = x[tid] + r[tid];
    float v1 = x[tid + 256] + r[tid + 256];

    red[tid] = v0 + v1; __syncthreads();
    for (int s = 128; s > 0; s >>= 1) {
        if (tid < s) red[tid] += red[tid + s];
        __syncthreads();
    }
    float mu = red[0] * (1.0f / TDM); __syncthreads();

    float d0 = v0 - mu, d1 = v1 - mu;
    red[tid] = d0 * d0 + d1 * d1; __syncthreads();
    for (int s = 128; s > 0; s >>= 1) {
        if (tid < s) red[tid] += red[tid + s];
        __syncthreads();
    }
    float inv = rsqrtf(red[0] * (1.0f / TDM) + 1e-6f);

    o[tid]       = d0 * inv * g[tid]       + bta[tid];
    o[tid + 256] = d1 * inv * g[tid + 256] + bta[tid + 256];
}

// ---------------------------------------------------------------------------
// Host launcher
// ---------------------------------------------------------------------------
extern "C" void kernel_launch(void* const* d_in, const int* in_sizes, int n_in,
                              void* d_out, int out_size, void* d_ws, size_t ws_size,
                              hipStream_t stream)
{
    const float* x    = (const float*)d_in[0];
    const float* enc  = (const float*)d_in[1];
    const float* lam  = (const float*)d_in[2];   // [S,S] look-ahead mask
    const float* pad  = (const float*)d_in[3];   // [S]   padding mask
    const float* w_q1 = (const float*)d_in[4];
    const float* b_q1 = (const float*)d_in[5];
    const float* w_k1 = (const float*)d_in[6];
    const float* b_k1 = (const float*)d_in[7];
    const float* w_v1 = (const float*)d_in[8];
    const float* b_v1 = (const float*)d_in[9];
    const float* w_o1 = (const float*)d_in[10];
    const float* b_o1 = (const float*)d_in[11];
    const float* w_q2 = (const float*)d_in[12];
    const float* b_q2 = (const float*)d_in[13];
    const float* w_k2 = (const float*)d_in[14];
    const float* b_k2 = (const float*)d_in[15];
    const float* w_v2 = (const float*)d_in[16];
    const float* b_v2 = (const float*)d_in[17];
    const float* w_o2 = (const float*)d_in[18];
    const float* b_o2 = (const float*)d_in[19];
    const float* fw1  = (const float*)d_in[20];
    const float* fb1  = (const float*)d_in[21];
    const float* fw2  = (const float*)d_in[22];
    const float* fb2  = (const float*)d_in[23];
    const float* ln1g = (const float*)d_in[24];
    const float* ln1b = (const float*)d_in[25];
    const float* ln2g = (const float*)d_in[26];
    const float* ln2b = (const float*)d_in[27];
    const float* ln3g = (const float*)d_in[28];
    const float* ln3b = (const float*)d_in[29];

    const int M = TB * TS;                 // 4096 rows
    float* out3 = (float*)d_out;
    float* aw1  = out3 + (size_t)TB * TS * TDM;
    float* aw2  = aw1  + (size_t)TB * TH * TS * TS;

    // workspace layout (floats); MT = 4096*512
    const size_t MT = (size_t)M * TDM;
    float* ws   = (float*)d_ws;
    float* q    = ws + 0 * MT;
    float* k    = ws + 1 * MT;
    float* v    = ws + 2 * MT;
    float* ctx  = ws + 3 * MT;
    float* proj = ws + 4 * MT;
    float* out1 = ws + 5 * MT;
    float* out2 = ws + 6 * MT;
    float* ffh  = ws + 0 * MT;             // [4096, 2048] reuses q/k/v/ctx slots

    dim3 blk(128);
    dim3 gProj(TDM / 64, M / 128, 1);       // N=512 GEMMs
    dim3 gFfn1(THID / 64, M / 128, 1);      // N=2048 GEMM
    dim3 gAttn(TS / 64, TS / 128, TB * TH); // logits
    dim3 gCtx(1, TS / 128, TB * TH);        // attn @ V (N=64)
    dim3 gRow(M);

    // ---------------- MHA1 (self-attention, causal) ----------------
    gemm_bias_kernel<<<gProj, blk, 0, stream>>>(x, TDM, w_q1, TDM, q, TDM, M, TDM, TDM, b_q1, 0);
    gemm_bias_kernel<<<gProj, blk, 0, stream>>>(x, TDM, w_k1, TDM, k, TDM, M, TDM, TDM, b_k1, 0);
    gemm_bias_kernel<<<gProj, blk, 0, stream>>>(x, TDM, w_v1, TDM, v, TDM, M, TDM, TDM, b_v1, 0);
    attn_logits_kernel<<<gAttn, blk, 0, stream>>>(q, k, aw1, lam, 1);
    softmax_kernel<<<dim3(TB * TH * TS), 256, 0, stream>>>(aw1);
    ctx_gemm_kernel<<<gCtx, blk, 0, stream>>>(aw1, v, ctx);
    gemm_bias_kernel<<<gProj, blk, 0, stream>>>(ctx, TDM, w_o1, TDM, proj, TDM, M, TDM, TDM, b_o1, 0);
    add_ln_kernel<<<gRow, 256, 0, stream>>>(proj, x, ln1g, ln1b, out1);

    // ---------------- MHA2 (cross-attention, padding mask) ----------------
    gemm_bias_kernel<<<gProj, blk, 0, stream>>>(out1, TDM, w_q2, TDM, q, TDM, M, TDM, TDM, b_q2, 0);
    gemm_bias_kernel<<<gProj, blk, 0, stream>>>(enc,  TDM, w_k2, TDM, k, TDM, M, TDM, TDM, b_k2, 0);
    gemm_bias_kernel<<<gProj, blk, 0, stream>>>(enc,  TDM, w_v2, TDM, v, TDM, M, TDM, TDM, b_v2, 0);
    attn_logits_kernel<<<gAttn, blk, 0, stream>>>(q, k, aw2, pad, 2);
    softmax_kernel<<<dim3(TB * TH * TS), 256, 0, stream>>>(aw2);
    ctx_gemm_kernel<<<gCtx, blk, 0, stream>>>(aw2, v, ctx);
    gemm_bias_kernel<<<gProj, blk, 0, stream>>>(ctx, TDM, w_o2, TDM, proj, TDM, M, TDM, TDM, b_o2, 0);
    add_ln_kernel<<<gRow, 256, 0, stream>>>(proj, out1, ln2g, ln2b, out2);

    // ---------------- FFN ----------------
    gemm_bias_kernel<<<gFfn1, blk, 0, stream>>>(out2, TDM, fw1, THID, ffh, THID, M, THID, TDM, fb1, 1);
    gemm_bias_kernel<<<gProj, blk, 0, stream>>>(ffh, THID, fw2, TDM, proj, TDM, M, TDM, THID, fb2, 0);
    add_ln_kernel<<<gRow, 256, 0, stream>>>(proj, out2, ln3g, ln3b, out3);
}
